// SAGEEncoder_87986700026316
// MI455X (gfx1250) — matrix operand order repaired and checked by reading, not compiled, not run on hardware
//
#include <hip/hip_runtime.h>

typedef __attribute__((ext_vector_type(16))) _Float16 v16h;
typedef __attribute__((ext_vector_type(8)))  _Float16 v8h;
typedef __attribute__((ext_vector_type(8)))  float    v8f;
typedef __attribute__((ext_vector_type(4)))  float    v4f;

#define DIM 128

// Convert one 128x128 f32 weight to f16, transposed: WT[n*128+k] = W[k*128+n].
// Transposed layout makes the WMMA B-operand per-lane fetch contiguous.
__global__ void sage_prep_w(const float* __restrict__ W, _Float16* __restrict__ WT) {
    int k = blockIdx.x;   // 0..127 (row of W)
    int n = threadIdx.x;  // 0..127 (col of W)
    WT[n * DIM + k] = (_Float16)W[k * DIM + n];
}

// Edge scatter: agg[dst] += h[src]; optionally deg[dst] += 1.
// 32 lanes per edge, 4 floats per lane (full 128-wide row).
__global__ __launch_bounds__(256) void sage_scatter(
    const float* __restrict__ h, const int* __restrict__ src,
    const int* __restrict__ dst, float* __restrict__ agg,
    float* __restrict__ deg, int n_edges, int add_deg)
{
    long long gid = (long long)blockIdx.x * blockDim.x + threadIdx.x;
    int e    = (int)(gid >> 5);
    int lane = (int)(gid & 31);
    if (e >= n_edges) return;
    int s = src[e];
    int d = dst[e];
    v4f v = *(const v4f*)(h + (long long)s * DIM + lane * 4);
    float* ap = agg + (long long)d * DIM + lane * 4;
    atomicAdd(ap + 0, v[0]);
    atomicAdd(ap + 1, v[1]);
    atomicAdd(ap + 2, v[2]);
    atomicAdd(ap + 3, v[3]);
    if (add_deg && lane == 0) atomicAdd(deg + d, 1.0f);
}

// One block = 16 rows x 128 cols of output; 8 waves, each wave owns a 16x16
// tile, K=128 for both GEMMs -> 8 x V_WMMA_F32_16X16X32_F16 per wave.
//
// Stage 1 (cooperative): 256 threads load the 16x128 A-tiles (agg scaled by
// 1/max(deg,1), and the root features h), convert f32->f16 ONCE, park in LDS.
// Stage 2: each wave feeds WMMA A-fragments from LDS (ds_load_b128), B from
// the transposed f16 weights in global (contiguous 32B per lane).
//
// out[i,:] = relu( (agg[i,:]/max(deg[i],1)) @ Wl + h[i,:] @ Wr + bias )
// `out` may alias `hroot` (layer 2): all hroot reads happen in stage 1 before
// the barrier; stores happen after it, so the alias is race-free.
__global__ __launch_bounds__(256) void sage_gemm(
    const float* __restrict__ agg, const float* __restrict__ deg,
    const float* __restrict__ hroot,
    const _Float16* __restrict__ WlT, const _Float16* __restrict__ WrT,
    const float* __restrict__ bias, float* __restrict__ out, int n_nodes)
{
    __shared__ _Float16 As[16 * DIM];   // scaled neighbor-mean tile, f16
    __shared__ _Float16 Hs[16 * DIM];   // root-feature tile, f16

    const int tid  = threadIdx.x;
    const int row0 = blockIdx.x * 16;

    // ---- Stage 1: cooperative load + convert (each thread: 1 row-chunk of 8) ----
    {
        const int r  = tid >> 4;         // 0..15  tile row
        const int cc = (tid & 15) * 8;   // 0..120 col base
        int grow = row0 + r;
        if (grow >= n_nodes) grow = n_nodes - 1;
        const float invr = 1.0f / fmaxf(deg[grow], 1.0f);

        const v4f* gp = (const v4f*)(agg + (long long)grow * DIM + cc);
        v4f u0 = gp[0], u1 = gp[1];
        const v4f* hp = (const v4f*)(hroot + (long long)grow * DIM + cc);
        v4f w0 = hp[0], w1 = hp[1];

        v8h pa, ph;
#pragma unroll
        for (int j = 0; j < 4; ++j) {
            pa[j]     = (_Float16)(u0[j] * invr);
            pa[4 + j] = (_Float16)(u1[j] * invr);
            ph[j]     = (_Float16)w0[j];
            ph[4 + j] = (_Float16)w1[j];
        }
        *(v8h*)&As[r * DIM + cc] = pa;
        *(v8h*)&Hs[r * DIM + cc] = ph;
    }
    __syncthreads();

    // ---- Stage 2: WMMA from LDS (A) and global f16 weights (B) ----
    const int lane = tid & 31;
    const int wave = tid >> 5;           // 0..7
    const int mlo  = lane & 15;
    const int hi   = lane >> 4;          // 0 or 1
    const int kb   = hi * 16;            // A-operand K base within 32-chunk
    const int colB = wave * 16 + mlo;

    v8f c = {};

    // (agg/deg) @ Wl
#pragma unroll
    for (int kc = 0; kc < DIM; kc += 32) {
        v8h lo = *(const v8h*)&As[mlo * DIM + kc + kb];
        v8h hh = *(const v8h*)&As[mlo * DIM + kc + kb + 8];
        v16h a;
#pragma unroll
        for (int j = 0; j < 8; ++j) { a[j] = lo[j]; a[8 + j] = hh[j]; }
        v16h b = *(const v16h*)(WlT + (long long)colB * DIM + kc + hi * 16);
        c = __builtin_amdgcn_wmma_f32_16x16x32_f16(false, a, false, b,
                                                   (short)0, c, false, false);
    }
    // h @ Wr
#pragma unroll
    for (int kc = 0; kc < DIM; kc += 32) {
        v8h lo = *(const v8h*)&Hs[mlo * DIM + kc + kb];
        v8h hh = *(const v8h*)&Hs[mlo * DIM + kc + kb + 8];
        v16h a;
#pragma unroll
        for (int j = 0; j < 8; ++j) { a[j] = lo[j]; a[8 + j] = hh[j]; }
        v16h b = *(const v16h*)(WrT + (long long)colB * DIM + kc + hi * 16);
        c = __builtin_amdgcn_wmma_f32_16x16x32_f16(false, a, false, b,
                                                   (short)0, c, false, false);
    }

    // ---- Epilogue: bias + ReLU; C/D layout row = r + 8*(lane>>4), col = lane&15 ----
    const float bv = bias[colB];
#pragma unroll
    for (int r = 0; r < 8; ++r) {
        int row = row0 + r + hi * 8;
        if (row < n_nodes) {
            float v = c[r] + bv;
            out[(long long)row * DIM + colB] = v > 0.0f ? v : 0.0f;
        }
    }
}

extern "C" void kernel_launch(void* const* d_in, const int* in_sizes, int n_in,
                              void* d_out, int out_size, void* d_ws, size_t ws_size,
                              hipStream_t stream) {
    (void)n_in; (void)out_size; (void)ws_size;

    const float* x   = (const float*)d_in[0];
    const int*   ei  = (const int*)d_in[1];
    const float* Wl1 = (const float*)d_in[2];
    const float* bl1 = (const float*)d_in[3];
    const float* Wr1 = (const float*)d_in[4];
    const float* Wl2 = (const float*)d_in[5];
    const float* bl2 = (const float*)d_in[6];
    const float* Wr2 = (const float*)d_in[7];

    const int n_nodes = in_sizes[0] / DIM;   // 100000
    const int n_edges = in_sizes[1] / 2;     // 1600000
    const int* src = ei;
    const int* dst = ei + n_edges;
    float* out = (float*)d_out;

    // Workspace layout: [deg][agg][WlT1][WrT1][WlT2][WrT2]
    char* ws = (char*)d_ws;
    size_t degBytes = (size_t)n_nodes * sizeof(float);
    size_t degPad   = (degBytes + 511) & ~(size_t)511;
    size_t aggBytes = (size_t)n_nodes * DIM * sizeof(float);
    size_t aggPad   = (aggBytes + 511) & ~(size_t)511;
    float*    deg   = (float*)ws;
    float*    agg   = (float*)(ws + degPad);
    char*     wbase = ws + degPad + aggPad;
    _Float16* WlT1  = (_Float16*)(wbase);
    _Float16* WrT1  = (_Float16*)(wbase + 32768);
    _Float16* WlT2  = (_Float16*)(wbase + 65536);
    _Float16* WrT2  = (_Float16*)(wbase + 98304);

    // Weights -> f16 transposed (cheap: 64 KB each; deterministic every call)
    sage_prep_w<<<DIM, DIM, 0, stream>>>(Wl1, WlT1);
    sage_prep_w<<<DIM, DIM, 0, stream>>>(Wr1, WrT1);
    sage_prep_w<<<DIM, DIM, 0, stream>>>(Wl2, WlT2);
    sage_prep_w<<<DIM, DIM, 0, stream>>>(Wr2, WrT2);

    // Zero deg + agg (contiguous region; memset node is graph-capturable)
    hipMemsetAsync(ws, 0, degPad + aggBytes, stream);

    const int sthreads = 256;
    long long total = (long long)n_edges * 32;
    const int sblocks = (int)((total + sthreads - 1) / sthreads);
    const int gblocks = (n_nodes + 15) / 16;

    // Layer 1: aggregate x, then fused dual-GEMM + bias + ReLU -> h1 (in d_out)
    sage_scatter<<<sblocks, sthreads, 0, stream>>>(x, src, dst, agg, deg, n_edges, 1);
    sage_gemm<<<gblocks, 256, 0, stream>>>(agg, deg, x, WlT1, WrT1, bl1, out, n_nodes);

    // Layer 2: aggregate h1 (from d_out), GEMM writes final result back to d_out
    hipMemsetAsync(agg, 0, aggBytes, stream);
    sage_scatter<<<sblocks, sthreads, 0, stream>>>(out, src, dst, agg, deg, n_edges, 0);
    sage_gemm<<<gblocks, 256, 0, stream>>>(agg, deg, out, WlT2, WrT2, bl2, out, n_nodes);
}
